// GAttention_11647951307130
// MI455X (gfx1250) — matrix operand order repaired and checked by reading, not compiled, not run on hardware
//
#include <hip/hip_runtime.h>
#include <hip/hip_bf16.h>

// GAttention for MI455X (gfx1250, wave32, WMMA).
//
// Pipeline (all bf16 WMMA with f32 accumulation):
//   1) pack_k : k[b,n,t] = sum_c alpha[c]*x[b,c,n,t], bf16, K padded 24->32
//   2) kg     : kG = k @ Gw, bf16, padded
//   3) pack_x : XT2[b][c*T+t][m] = x[b,c,m,t] in bf16 (transposed for GEMM A)
//   4) attn   : per (b, 16-row block): scores via v_wmma_f32_16x16x32_bf16,
//               lane-private online softmax (merge once at end), W = att*adj bf16
//   5) gemm   : out[b, j=(c,t), n] = sum_m XT2[j,m] * W[n,m], 2x2 register
//               tiling (4 WMMAs / 4 frag loads), f32 accumulate, prefetch
//
// Workspace layout (228 MiB total):
//   [0,   2MiB)  kpad  bf16 [16][2048][32]
//   [2,   4MiB)  kGpad bf16 [16][2048][32]
//   [4, 100MiB)  XT2   bf16 [16][1536][2048]
//   [100,228MiB) W     bf16 [16][2048][2048]

typedef __attribute__((ext_vector_type(16))) __bf16 v16bf;
typedef __attribute__((ext_vector_type(8)))  __bf16 v8bf;
typedef __attribute__((ext_vector_type(8)))  float  v8f;

#define NB   2048
#define CB   64
#define TB   24
#define BB   16
#define JDIM (CB * TB)   // 1536

// ---- WMMA fragment loaders using contiguous k order for A and B ------------------
// A (16x32 bf16, row-major source, row stride in elements):
//   lane L<16 : row L,      K = k0+0..15
//   lane L>=16: row L-16,   K = k0+16..31
__device__ __forceinline__ v16bf load_frag_a(const __bf16* __restrict__ base,
                                             int row0, int k0, int stride, int lane) {
    int r = lane & 15, h = lane >> 4;
    const __bf16* p = base + (size_t)(row0 + r) * stride + k0 + h * 16;
    v8bf lo = *(const v8bf*)(p);
    v8bf hi = *(const v8bf*)(p + 8);
    return __builtin_shufflevector(lo, hi, 0,1,2,3,4,5,6,7,8,9,10,11,12,13,14,15);
}

// B (32x16 bf16). Element (K, col) = S[col0+col][k0+K], S row-major.
//   lane L<16 : col L,    K = k0+0..15   (contiguous 32B)
//   lane L>=16: col L-16, K = k0+16..31  (contiguous 32B)
__device__ __forceinline__ v16bf load_frag_b(const __bf16* __restrict__ base,
                                             int col0, int k0, int stride, int lane) {
    int c = lane & 15, h = lane >> 4;
    const __bf16* p = base + (size_t)(col0 + c) * stride + k0 + h * 16;
    return *(const v16bf*)(p);
}

// C/D (16x16 f32): lane L<16 -> (M=v, N=L); lane L>=16 -> (M=v+8, N=L-16).

// ---- 1) k reduction ---------------------------------------------------------
__global__ void __launch_bounds__(256)
pack_k_kernel(const float* __restrict__ x, const float* __restrict__ alpha,
              __bf16* __restrict__ kpad) {
    int idx = blockIdx.x * 256 + threadIdx.x;       // 16*2048*32 threads
    int s = idx & 31;
    int n = (idx >> 5) & (NB - 1);
    int b = idx >> 16;
    float acc = 0.0f;
    if (s < TB) {
        const float* xp = x + ((size_t)b * CB * NB + n) * TB + s;
        #pragma unroll 4
        for (int c = 0; c < CB; ++c)
            acc += xp[(size_t)c * NB * TB] * alpha[c];
    }
    kpad[idx] = (__bf16)acc;
}

// ---- 2) kG = k @ Gw ---------------------------------------------------------
__global__ void __launch_bounds__(256)
kg_kernel(const __bf16* __restrict__ kpad, const float* __restrict__ Gw,
          __bf16* __restrict__ kG) {
    int idx = blockIdx.x * 256 + threadIdx.x;       // 16*2048*32 threads
    int s = idx & 31;
    float acc = 0.0f;
    if (s < TB) {
        const __bf16* kp = kpad + (idx & ~31);      // row base
        #pragma unroll
        for (int t = 0; t < TB; ++t)
            acc += (float)kp[t] * Gw[t * TB + s];
    }
    kG[idx] = (__bf16)acc;
}

// ---- 3) transpose pack of x -------------------------------------------------
__global__ void __launch_bounds__(256)
pack_x_kernel(const float* __restrict__ x, __bf16* __restrict__ XT2) {
    size_t idx = (size_t)blockIdx.x * 256 + threadIdx.x;   // 16*1536*2048 threads
    int m = (int)(idx & (NB - 1));
    int j = (int)((idx >> 11) % JDIM);
    int b = (int)(idx / ((size_t)JDIM * NB));
    int c = j / TB, t = j - c * TB;
    XT2[idx] = (__bf16)x[(((size_t)b * CB + c) * NB + m) * TB + t];
}

// ---- 4) attention: scores WMMA + online softmax + adj mask ------------------
__global__ void __launch_bounds__(128)
attn_kernel(const __bf16* __restrict__ kpad, const __bf16* __restrict__ kG,
            const float* __restrict__ adj, __bf16* __restrict__ W) {
    int b    = blockIdx.x >> 7;          // / 128
    int n0   = (blockIdx.x & 127) * 16;
    int lane = threadIdx.x & 31;
    int wave = threadIdx.x >> 5;
    int half = lane >> 4;

    const __bf16* kb = kpad + (size_t)b * NB * 32;
    const __bf16* gb = kG   + (size_t)b * NB * 32;

    // A fragment: kG rows n0..n0+15, full K=32 (zero padded past 24)
    v16bf a = load_frag_a(gb, n0, 0, 32, lane);

    // Lane-private online softmax state: each lane owns one column per tile,
    // so no cross-lane traffic inside the tile loop.
    float mrun[8], srun[8];
    #pragma unroll
    for (int v = 0; v < 8; ++v) { mrun[v] = -3.0e38f; srun[v] = 0.0f; }

    // Pass 1: per-lane online max/sum over this wave's 32 column tiles
    for (int mt = wave; mt < 128; mt += 4) {
        v16bf bf = load_frag_b(kb, mt * 16, 0, 32, lane);
        v8f d = {};
        d = __builtin_amdgcn_wmma_f32_16x16x32_bf16(false, a, false, bf,
                                                    (short)0, d, false, false);
        #pragma unroll
        for (int v = 0; v < 8; ++v) {
            float xv = d[v];
            float mo = mrun[v];
            float mn = fmaxf(mo, xv);
            srun[v] = srun[v] * __expf(mo - mn) + __expf(xv - mn);
            mrun[v] = mn;
        }
    }

    // Merge (max,sum) across the 16 lanes of each half: 4 log-sum-exp steps
    #pragma unroll
    for (int off = 1; off <= 8; off <<= 1) {
        #pragma unroll
        for (int v = 0; v < 8; ++v) {
            float mo = __shfl_xor(mrun[v], off, 32);
            float so = __shfl_xor(srun[v], off, 32);
            float mn = fmaxf(mrun[v], mo);
            srun[v] = srun[v] * __expf(mrun[v] - mn) + so * __expf(mo - mn);
            mrun[v] = mn;
        }
    }

    // Combine per-wave stats across the 4 waves
    __shared__ float sm[4][16], ss[4][16], gM[16], gInv[16];
    if ((lane & 15) == 0) {
        #pragma unroll
        for (int v = 0; v < 8; ++v) {
            sm[wave][v + 8 * half] = mrun[v];
            ss[wave][v + 8 * half] = srun[v];
        }
    }
    __syncthreads();
    if (threadIdx.x < 16) {
        int r = threadIdx.x;
        float mg = fmaxf(fmaxf(sm[0][r], sm[1][r]), fmaxf(sm[2][r], sm[3][r]));
        float sg = ss[0][r] * __expf(sm[0][r] - mg) + ss[1][r] * __expf(sm[1][r] - mg)
                 + ss[2][r] * __expf(sm[2][r] - mg) + ss[3][r] * __expf(sm[3][r] - mg);
        gM[r]   = mg;
        gInv[r] = 1.0f / sg;
    }
    __syncthreads();

    float gm[8], gi[8];
    #pragma unroll
    for (int v = 0; v < 8; ++v) { gm[v] = gM[v + 8 * half]; gi[v] = gInv[v + 8 * half]; }

    // Pass 2: recompute tiles, emit W = softmax(scores)*adj in bf16
    __bf16* Wb = W + (size_t)b * NB * NB;
    int col = lane & 15;
    for (int mt = wave; mt < 128; mt += 4) {
        int m0 = mt * 16;
        v16bf bf = load_frag_b(kb, m0, 0, 32, lane);
        v8f d = {};
        d = __builtin_amdgcn_wmma_f32_16x16x32_bf16(false, a, false, bf,
                                                    (short)0, d, false, false);
        #pragma unroll
        for (int v = 0; v < 8; ++v) {
            int n = n0 + v + 8 * half;
            float w = __expf(d[v] - gm[v]) * gi[v] * adj[(size_t)n * NB + m0 + col];
            Wb[(size_t)n * NB + m0 + col] = (__bf16)w;
        }
    }
}

// ---- 5) out[j,n] = sum_m XT2[j,m] * W[n,m]  (the 206 GFLOP GEMM) -----------
// One wave computes a 32x32 output block (2x2 WMMA tiles): 4 WMMAs per
// 2 A-frag + 2 B-frag loads -> half the L2 traffic per FLOP vs 1x1.
__global__ void __launch_bounds__(256)
out_gemm_kernel(const __bf16* __restrict__ XT2, const __bf16* __restrict__ W,
                float* __restrict__ out) {
    int gid  = blockIdx.x * 8 + (threadIdx.x >> 5);  // one wave per 32x32 block
    int lane = threadIdx.x & 31;
    int nt2  = gid & 63;                             // 2048/32
    int rest = gid >> 6;
    int jt2  = rest % (JDIM / 32);                   // 48
    int b    = rest / (JDIM / 32);
    int j0 = jt2 * 32, n0 = nt2 * 32;

    const __bf16* Xb = XT2 + (size_t)b * JDIM * NB;
    const __bf16* Wb = W   + (size_t)b * NB * NB;

    v8f acc00 = {}, acc01 = {}, acc10 = {}, acc11 = {};
    for (int kt = 0; kt < NB / 32; ++kt) {
        int k0 = kt * 32;
        v16bf a0 = load_frag_a(Xb, j0,      k0, NB, lane);
        v16bf a1 = load_frag_a(Xb, j0 + 16, k0, NB, lane);
        v16bf b0 = load_frag_b(Wb, n0,      k0, NB, lane);
        v16bf b1 = load_frag_b(Wb, n0 + 16, k0, NB, lane);
        acc00 = __builtin_amdgcn_wmma_f32_16x16x32_bf16(false, a0, false, b0,
                                                        (short)0, acc00, false, false);
        acc01 = __builtin_amdgcn_wmma_f32_16x16x32_bf16(false, a0, false, b1,
                                                        (short)0, acc01, false, false);
        acc10 = __builtin_amdgcn_wmma_f32_16x16x32_bf16(false, a1, false, b0,
                                                        (short)0, acc10, false, false);
        acc11 = __builtin_amdgcn_wmma_f32_16x16x32_bf16(false, a1, false, b1,
                                                        (short)0, acc11, false, false);
        if (kt + 1 < NB / 32) {
            // 32 lanes cover the 32 rows of each operand's next K slice
            __builtin_prefetch(Xb + (size_t)(j0 + lane) * NB + k0 + 32, 0, 0);
            __builtin_prefetch(Wb + (size_t)(n0 + lane) * NB + k0 + 32, 0, 0);
        }
    }

    int half = lane >> 4, col = lane & 15;
    #pragma unroll
    for (int tj = 0; tj < 2; ++tj) {
        v8f accn0 = tj ? acc10 : acc00;
        v8f accn1 = tj ? acc11 : acc01;
        #pragma unroll
        for (int v = 0; v < 8; ++v) {
            int j = j0 + tj * 16 + v + 8 * half;
            int c = j / TB, t = j - c * TB;
            size_t base = ((size_t)b * CB + c) * NB;
            out[(base + n0 + col) * TB + t]      = accn0[v];
            out[(base + n0 + 16 + col) * TB + t] = accn1[v];
        }
    }
}

// ---- launch -----------------------------------------------------------------
extern "C" void kernel_launch(void* const* d_in, const int* in_sizes, int n_in,
                              void* d_out, int out_size, void* d_ws, size_t ws_size,
                              hipStream_t stream) {
    const float* x     = (const float*)d_in[0];
    const float* adj   = (const float*)d_in[1];
    const float* Gw    = (const float*)d_in[2];
    const float* alpha = (const float*)d_in[3];
    float* out = (float*)d_out;

    char* ws = (char*)d_ws;
    __bf16* kpad = (__bf16*)(ws);
    __bf16* kG   = (__bf16*)(ws + (size_t)2 * 1024 * 1024);
    __bf16* XT2  = (__bf16*)(ws + (size_t)4 * 1024 * 1024);
    __bf16* W    = (__bf16*)(ws + (size_t)4 * 1024 * 1024
                                + (size_t)BB * JDIM * NB * sizeof(__bf16));
    // total workspace use: 228 MiB

    pack_k_kernel<<<(BB * NB * 32) / 256, 256, 0, stream>>>(x, alpha, kpad);
    kg_kernel<<<(BB * NB * 32) / 256, 256, 0, stream>>>(kpad, Gw, kG);
    pack_x_kernel<<<(unsigned)(((size_t)BB * JDIM * NB) / 256), 256, 0, stream>>>(x, XT2);
    attn_kernel<<<BB * (NB / 16), 128, 0, stream>>>(kpad, kG, adj, W);
    // 16 * 48 * 64 = 49152 waves, 8 per block
    out_gemm_kernel<<<(BB * (JDIM / 32) * (NB / 32)) / 8, 256, 0, stream>>>(XT2, W, out);
}
